// TransformerEncoderBlock_58566174049022
// MI455X (gfx1250) — matrix operand order repaired and checked by reading, not compiled, not run on hardware
//
#include <hip/hip_runtime.h>

// ---------------- problem constants ----------------
#define Bc  4
#define Sc  2048
#define Dc  512
#define Hc  8
#define DHc 64
#define NROWS (Bc * Sc)          // 8192
#define NEGBIG (-1e9f)

typedef __attribute__((ext_vector_type(16))) __bf16 v16bf;
typedef __attribute__((ext_vector_type(8)))  __bf16 v8bf;
typedef __attribute__((ext_vector_type(8)))  float  v8f;
typedef __attribute__((ext_vector_type(4)))  unsigned int u32x4;
typedef __attribute__((ext_vector_type(8)))  int i32x8;
typedef __attribute__((ext_vector_type(4)))  int i32x4;

#ifndef __has_builtin
#define __has_builtin(x) 0
#endif
#if __has_builtin(__builtin_amdgcn_tensor_load_to_lds) && \
    __has_builtin(__builtin_amdgcn_s_wait_tensorcnt)
#define HAVE_TDM 1
#else
#define HAVE_TDM 0
#endif

#define WMMA_BF16(a, b, c) \
  __builtin_amdgcn_wmma_f32_16x16x32_bf16(false, (a), false, (b), (short)0, (c), false, false)

__device__ __forceinline__ v16bf make_v16(v8bf lo, v8bf hi) {
  return __builtin_shufflevector(lo, hi, 0, 1, 2, 3, 4, 5, 6, 7, 8, 9, 10, 11, 12, 13, 14, 15);
}

// A-fragment (16x32 bf16): lane m=lane&15 holds row m; one contiguous 16-elem K-run (k = 16*(lane>>4) + s)
__device__ __forceinline__ v16bf load_a_frag(const __bf16* lds, int row0, int ldm, int lane) {
  int m = lane & 15, kh = lane >> 4;
  const __bf16* p = lds + (size_t)(row0 + m) * ldm;
  v8bf lo = *(const v8bf*)(p + 16 * kh);
  v8bf hi = *(const v8bf*)(p + 16 * kh + 8);
  return make_v16(lo, hi);
}
// B-fragment from K-contiguous (transposed) tile: one contiguous 16-elem K-run
__device__ __forceinline__ v16bf load_bt_frag(const __bf16* lds, int col0, int ldm, int lane) {
  int n = lane & 15, kh = lane >> 4;
  const __bf16* p = lds + (size_t)(col0 + n) * ldm + 16 * kh;
  v8bf lo = *(const v8bf*)(p);
  v8bf hi = *(const v8bf*)(p + 8);
  return make_v16(lo, hi);
}

#if HAVE_TDM
// ---------------- Tensor Data Mover: 2D bf16 tile load (ISA ch.8 D# layout) ----------------
__device__ __forceinline__ void tdm_load_2d_bf16(unsigned lds_addr, const void* gaddr,
                                                 unsigned tensor_d0, unsigned tensor_d1,
                                                 unsigned stride0,
                                                 unsigned tile_d0, unsigned tile_d1,
                                                 unsigned pad_interval_code,
                                                 unsigned pad_amount_code) {
  unsigned long long ga = (unsigned long long)(size_t)gaddr;
  u32x4 g0;
  g0[0] = 1u;                                            // count=1 (valid user D#)
  g0[1] = lds_addr;                                      // lds_addr [63:32]
  g0[2] = (unsigned)(ga & 0xFFFFFFFFu);                  // global_addr [95:64]
  g0[3] = (unsigned)((ga >> 32) & 0x1FFFFFFu) | (2u << 30);  // addr msbs + type=2
  i32x8 g1;
  g1[0] = (int)((1u << 16) | (1u << 20) |                // data_size=2B, pad_enable
                (pad_interval_code << 22) | (pad_amount_code << 25));
  g1[1] = (int)((tensor_d0 & 0xFFFFu) << 16);            // tensor_dim0[15:0]
  g1[2] = (int)(((tensor_d0 >> 16) & 0xFFFFu) | ((tensor_d1 & 0xFFFFu) << 16));
  g1[3] = (int)(((tensor_d1 >> 16) & 0xFFFFu) | ((tile_d0 & 0xFFFFu) << 16));
  g1[4] = (int)(tile_d1 & 0xFFFFu);                      // tile_dim1; tile_dim2=0
  g1[5] = (int)stride0;                                  // tensor_dim0_stride[31:0]
  g1[6] = 0;
  g1[7] = 0;
  i32x4 gz = {0, 0, 0, 0};
#if __clang_major__ >= 23
  i32x8 gz8 = {0, 0, 0, 0, 0, 0, 0, 0};
  __builtin_amdgcn_tensor_load_to_lds(g0, g1, gz, gz, gz8, 0);
#else
  __builtin_amdgcn_tensor_load_to_lds(g0, g1, gz, gz, 0);
#endif
}
#endif  // HAVE_TDM

// ---------------- conversion / packing kernels ----------------
__global__ __launch_bounds__(256) void f32_to_bf16(const float* __restrict__ in,
                                                   __bf16* __restrict__ out, int n) {
  int i = blockIdx.x * 256 + threadIdx.x;
  if (i < n) out[i] = (__bf16)in[i];
}

__global__ __launch_bounds__(256) void transpose_to_bf16(const float* __restrict__ in,
                                                         __bf16* __restrict__ out) {
  int i = blockIdx.x * 256 + threadIdx.x;   // over D*D
  if (i < Dc * Dc) {
    int k = i >> 9, n = i & (Dc - 1);
    out[(size_t)n * Dc + k] = (__bf16)in[i];
  }
}

__global__ __launch_bounds__(256) void pack_qkv_w(const float* __restrict__ w,
                                                  __bf16* __restrict__ out) {
  int i = blockIdx.x * 256 + threadIdx.x;   // over H*D*DH
  if (i < Hc * Dc * DHc) {
    int h = i / (Dc * DHc);
    int r = i - h * Dc * DHc;
    int d = r / DHc;
    int e = r - d * DHc;
    out[(size_t)(h * DHc + e) * Dc + d] = (__bf16)w[i];
  }
}

__global__ __launch_bounds__(256) void token_bias(const long long* __restrict__ tok,
                                                  float* __restrict__ biasOut, int n) {
  int i = blockIdx.x * 256 + threadIdx.x;
  if (i < n) biasOut[i] = (tok[i] != 0) ? 0.f : NEGBIG;
}

// ---------------- tiled bf16 WMMA GEMM (double-buffered TDM-fed LDS tiles) ----------------
// C[M,N] = A[M,K] @ Bt[N,K]^T; block tile 64(M) x 128(N), K-step 32, 128 thr (4 waves)
__global__ __launch_bounds__(128) void gemm_bf16(const __bf16* __restrict__ A,
                                                 const __bf16* __restrict__ Bt,
                                                 int M, int N, int K, int mode,
                                                 const float* __restrict__ resid,
                                                 const float* __restrict__ bias,
                                                 void* __restrict__ out) {
  __shared__ __align__(16) __bf16 ldsA[2][64][40];     // rows of 32 K + 8 pad
  __shared__ __align__(16) __bf16 ldsBT[2][128][40];
  const int tid  = threadIdx.x;
  const int lane = tid & 31, wave = tid >> 5;
  const int wm = wave >> 1, wn = wave & 1;
  const int m0 = blockIdx.y * 64, n0 = blockIdx.x * 128;
  const int NIT = K >> 5;

  v8f zero = {};
  v8f acc[2][4];
#pragma unroll
  for (int i = 0; i < 2; ++i)
#pragma unroll
    for (int j = 0; j < 4; ++j) acc[i][j] = zero;

#if HAVE_TDM
  if (tid == 0) {   // prime buffer 0 (tile k0 = 0)
    tdm_load_2d_bf16((unsigned)(size_t)&ldsA[0][0][0], &A[(size_t)m0 * K],
                     (unsigned)K, (unsigned)M, (unsigned)K, 32u, 64u, 3u, 3u);
    tdm_load_2d_bf16((unsigned)(size_t)&ldsBT[0][0][0], &Bt[(size_t)n0 * K],
                     (unsigned)K, (unsigned)N, (unsigned)K, 32u, 128u, 3u, 3u);
  }
#endif
  for (int it = 0; it < NIT; ++it) {
    const int cur = it & 1;
    const int k0 = it << 5;
#if HAVE_TDM
    if (tid == 0) {
      if (it + 1 < NIT) {   // DMA next tile into the other buffer, then wait only for cur
        int kn = k0 + 32;
        tdm_load_2d_bf16((unsigned)(size_t)&ldsA[cur ^ 1][0][0], &A[(size_t)m0 * K + kn],
                         (unsigned)K, (unsigned)M, (unsigned)K, 32u, 64u, 3u, 3u);
        tdm_load_2d_bf16((unsigned)(size_t)&ldsBT[cur ^ 1][0][0], &Bt[(size_t)n0 * K + kn],
                         (unsigned)K, (unsigned)N, (unsigned)K, 32u, 128u, 3u, 3u);
        __builtin_amdgcn_s_wait_tensorcnt((short)2);
      } else {
        __builtin_amdgcn_s_wait_tensorcnt((short)0);
      }
    }
#else
#pragma unroll
    for (int i = tid; i < 256; i += 128) {
      int r = i >> 2, c = (i & 3) * 8;
      *(v8bf*)&ldsA[cur][r][c] = *(const v8bf*)&A[(size_t)(m0 + r) * K + k0 + c];
    }
#pragma unroll
    for (int i = tid; i < 512; i += 128) {
      int r = i >> 2, c = (i & 3) * 8;
      *(v8bf*)&ldsBT[cur][r][c] = *(const v8bf*)&Bt[(size_t)(n0 + r) * K + k0 + c];
    }
#endif
    __syncthreads();

    v16bf a0 = load_a_frag(&ldsA[cur][0][0], 32 * wm,      40, lane);
    v16bf a1 = load_a_frag(&ldsA[cur][0][0], 32 * wm + 16, 40, lane);
    v16bf b[4];
#pragma unroll
    for (int j = 0; j < 4; ++j)
      b[j] = load_bt_frag(&ldsBT[cur][0][0], 64 * wn + 16 * j, 40, lane);

#pragma unroll
    for (int j = 0; j < 4; ++j) {
      acc[0][j] = WMMA_BF16(a0, b[j], acc[0][j]);
      acc[1][j] = WMMA_BF16(a1, b[j], acc[1][j]);
    }
    __syncthreads();   // all reads of buf[cur] done before it is re-DMA'd at it+2
  }

  const int kh = lane >> 4, nl = lane & 15;
#pragma unroll
  for (int ti = 0; ti < 2; ++ti)
#pragma unroll
    for (int tj = 0; tj < 4; ++tj)
#pragma unroll
      for (int r = 0; r < 8; ++r) {
        int gm = m0 + 32 * wm + 16 * ti + r + 8 * kh;
        int gn = n0 + 64 * wn + 16 * tj + nl;
        float val = acc[ti][tj][r];
        if (mode == 0) {
          __bf16* o = (__bf16*)out;
          int bb = gm >> 11, ss = gm & (Sc - 1);
          int hh = gn >> 6,  ee = gn & (DHc - 1);
          o[(((size_t)bb * Hc + hh) * Sc + ss) * DHc + ee] = (__bf16)val;
        } else if (mode == 1) {
          float* o = (float*)out;
          size_t idx = (size_t)gm * N + gn;
          o[idx] = val + resid[idx];
        } else if (mode == 2) {
          __bf16* o = (__bf16*)out;
          float t = val + bias[gn];
          o[(size_t)gm * N + gn] = (__bf16)(t > 0.f ? t : 0.f);
        } else if (mode == 3) {
          float* o = (float*)out;
          o[(size_t)gm * N + gn] = val + bias[gn];
        } else {  // mode 4: V transposed [B,H,DH,S]
          __bf16* o = (__bf16*)out;
          int bb = gm >> 11, ss = gm & (Sc - 1);
          int hh = gn >> 6,  ee = gn & (DHc - 1);
          o[(((size_t)bb * Hc + hh) * DHc + ee) * Sc + ss] = (__bf16)val;
        }
      }
}

// ------------- flash attention (streaming softmax, double-buffered TDM K/V) -------------
__global__ __launch_bounds__(128) void flash_attn(const __bf16* __restrict__ q,
                                                  const __bf16* __restrict__ k,
                                                  const __bf16* __restrict__ vT,
                                                  const float* __restrict__ maskBias,
                                                  __bf16* __restrict__ attn_out) {
  __shared__ __align__(16) __bf16 ldsK[2][32][72];    // 32 keys x 64 dh (+8 pad)
  __shared__ __align__(16) __bf16 ldsVT[2][64][40];   // 64 dh x 32 keys (+8 pad)
  __shared__ __align__(16) __bf16 pBuf[4][16][40];

  const int tid = threadIdx.x, lane = tid & 31, wave = tid >> 5;
  const int qt = blockIdx.x & 31;
  const int bh = blockIdx.x >> 5;
  const int b = bh >> 3, h = bh & 7;
  const int nl = lane & 15, kh = lane >> 4;
  const size_t headOff = (size_t)(b * Hc + h) * Sc * DHc;
  const int q0 = qt * 64 + wave * 16;
  const int NIT = Sc / 32;

  v16bf qf[2];
  {
    const __bf16* qp = q + headOff + (size_t)(q0 + nl) * DHc;
#pragma unroll
    for (int f = 0; f < 2; ++f) {
      v8bf lo = *(const v8bf*)(qp + 32 * f + 16 * kh);
      v8bf hi = *(const v8bf*)(qp + 32 * f + 16 * kh + 8);
      qf[f] = make_v16(lo, hi);
    }
  }

  v8f zero = {};
  v8f acc[4];
#pragma unroll
  for (int c = 0; c < 4; ++c) acc[c] = zero;
  float mrow[8], lrow[8];
#pragma unroll
  for (int r = 0; r < 8; ++r) { mrow[r] = -3.0e38f; lrow[r] = 0.f; }

#if HAVE_TDM
  if (tid == 0) {   // prime buffer 0 (keys j0 = 0)
    tdm_load_2d_bf16((unsigned)(size_t)&ldsK[0][0][0], k + headOff,
                     (unsigned)DHc, (unsigned)Sc, (unsigned)DHc, 64u, 32u, 4u, 3u);
    tdm_load_2d_bf16((unsigned)(size_t)&ldsVT[0][0][0], vT + headOff,
                     (unsigned)Sc, (unsigned)DHc, (unsigned)Sc, 32u, 64u, 3u, 3u);
  }
#endif
  for (int it = 0; it < NIT; ++it) {
    const int cur = it & 1;
    const int j0 = it * 32;
#if HAVE_TDM
    if (tid == 0) {
      if (it + 1 < NIT) {
        int jn = j0 + 32;
        tdm_load_2d_bf16((unsigned)(size_t)&ldsK[cur ^ 1][0][0],
                         k + headOff + (size_t)jn * DHc,
                         (unsigned)DHc, (unsigned)Sc, (unsigned)DHc, 64u, 32u, 4u, 3u);
        tdm_load_2d_bf16((unsigned)(size_t)&ldsVT[cur ^ 1][0][0],
                         vT + headOff + (size_t)jn,
                         (unsigned)Sc, (unsigned)DHc, (unsigned)Sc, 32u, 64u, 3u, 3u);
        __builtin_amdgcn_s_wait_tensorcnt((short)2);
      } else {
        __builtin_amdgcn_s_wait_tensorcnt((short)0);
      }
    }
#else
    {
      const __bf16* kb  = k  + headOff + (size_t)j0 * DHc;
      const __bf16* vtb = vT + headOff + (size_t)j0;
#pragma unroll
      for (int i = tid; i < 256; i += 128) {
        int r = i >> 3, c = (i & 7) * 8;
        *(v8bf*)&ldsK[cur][r][c] = *(const v8bf*)&kb[(size_t)r * DHc + c];
      }
#pragma unroll
      for (int i = tid; i < 256; i += 128) {
        int r = i >> 2, c = (i & 3) * 8;
        *(v8bf*)&ldsVT[cur][r][c] = *(const v8bf*)&vtb[(size_t)r * Sc + c];
      }
    }
#endif
    __syncthreads();

    v8f sc[2];
    sc[0] = zero; sc[1] = zero;
#pragma unroll
    for (int t = 0; t < 2; ++t)
#pragma unroll
      for (int f = 0; f < 2; ++f) {
        const __bf16* p = &ldsK[cur][16 * t + nl][32 * f + 16 * kh];
        v16bf bf_ = make_v16(*(const v8bf*)p, *(const v8bf*)(p + 8));
        sc[t] = WMMA_BF16(qf[f], bf_, sc[t]);
      }

    float bias0 = maskBias[(size_t)b * Sc + j0 + nl];
    float bias1 = maskBias[(size_t)b * Sc + j0 + 16 + nl];

    float so[8];
#pragma unroll
    for (int r = 0; r < 8; ++r) {
      float s0 = sc[0][r] * 0.125f + bias0;
      float s1 = sc[1][r] * 0.125f + bias1;
      float mx = fmaxf(s0, s1);
#pragma unroll
      for (int off = 1; off < 16; off <<= 1) mx = fmaxf(mx, __shfl_xor(mx, off, 32));
      float mnew = fmaxf(mrow[r], mx);
      float p0 = __expf(s0 - mnew), p1 = __expf(s1 - mnew);
      sc[0][r] = p0; sc[1][r] = p1;
      float sold = __expf(mrow[r] - mnew);
      float rs = p0 + p1;
#pragma unroll
      for (int off = 1; off < 16; off <<= 1) rs += __shfl_xor(rs, off, 32);
      lrow[r] = lrow[r] * sold + rs;
      mrow[r] = mnew;
      so[r] = sold;
    }
#pragma unroll
    for (int c = 0; c < 4; ++c)
#pragma unroll
      for (int r = 0; r < 8; ++r) acc[c][r] *= so[r];

#pragma unroll
    for (int t = 0; t < 2; ++t)
#pragma unroll
      for (int r = 0; r < 8; ++r)
        pBuf[wave][r + 8 * kh][16 * t + nl] = (__bf16)sc[t][r];
    v16bf pf;
    {
      const __bf16* p = &pBuf[wave][nl][0];
      v8bf lo = *(const v8bf*)(p + 16 * kh);
      v8bf hi = *(const v8bf*)(p + 16 * kh + 8);
      pf = make_v16(lo, hi);
    }
#pragma unroll
    for (int c = 0; c < 4; ++c) {
      const __bf16* p = &ldsVT[cur][16 * c + nl][16 * kh];
      v16bf vf = make_v16(*(const v8bf*)p, *(const v8bf*)(p + 8));
      acc[c] = WMMA_BF16(pf, vf, acc[c]);
    }
    __syncthreads();   // reads of buf[cur] done before its re-DMA at it+2
  }

#pragma unroll
  for (int c = 0; c < 4; ++c)
#pragma unroll
    for (int r = 0; r < 8; ++r) {
      int ss = q0 + r + 8 * kh;
      int dcol = h * DHc + 16 * c + nl;
      float o = acc[c][r] / lrow[r];
      attn_out[((size_t)b * Sc + ss) * Dc + dcol] = (__bf16)o;
    }
}

// ---------------- LayerNorm + residual ----------------
__global__ __launch_bounds__(256) void ln_residual(const float* __restrict__ y,
                                                   const float* __restrict__ resid,
                                                   const float* __restrict__ g,
                                                   const float* __restrict__ beta,
                                                   float* __restrict__ outf,
                                                   __bf16* __restrict__ outb) {
  __shared__ float red[8];
  const int row = blockIdx.x, tid = threadIdx.x;
  const size_t base = (size_t)row * Dc;
  float v0 = y[base + tid], v1 = y[base + tid + 256];
  float s = v0 + v1;
#pragma unroll
  for (int off = 16; off > 0; off >>= 1) s += __shfl_xor(s, off, 32);
  if ((tid & 31) == 0) red[tid >> 5] = s;
  __syncthreads();
  float tot = 0.f;
#pragma unroll
  for (int i = 0; i < 8; ++i) tot += red[i];
  float mean = tot * (1.0f / Dc);
  __syncthreads();
  float d0 = v0 - mean, d1 = v1 - mean;
  s = d0 * d0 + d1 * d1;
#pragma unroll
  for (int off = 16; off > 0; off >>= 1) s += __shfl_xor(s, off, 32);
  if ((tid & 31) == 0) red[tid >> 5] = s;
  __syncthreads();
  tot = 0.f;
#pragma unroll
  for (int i = 0; i < 8; ++i) tot += red[i];
  float inv = rsqrtf(tot * (1.0f / Dc) + 1e-5f);
  float o0 = d0 * inv * g[tid]       + beta[tid]       + resid[base + tid];
  float o1 = d1 * inv * g[tid + 256] + beta[tid + 256] + resid[base + tid + 256];
  outf[base + tid] = o0;
  outf[base + tid + 256] = o1;
  if (outb) {
    outb[base + tid] = (__bf16)o0;
    outb[base + tid + 256] = (__bf16)o1;
  }
}

// ---------------- host-side orchestration ----------------
extern "C" void kernel_launch(void* const* d_in, const int* in_sizes, int n_in,
                              void* d_out, int out_size, void* d_ws, size_t ws_size,
                              hipStream_t stream) {
  (void)in_sizes; (void)n_in; (void)out_size; (void)ws_size;
  const float*     x      = (const float*)d_in[0];
  const long long* tokens = (const long long*)d_in[1];
  const float* Wq = (const float*)d_in[2];
  const float* Wk = (const float*)d_in[3];
  const float* Wv = (const float*)d_in[4];
  const float* Wo = (const float*)d_in[5];
  const float* W1 = (const float*)d_in[6];
  const float* b1 = (const float*)d_in[7];
  const float* W2 = (const float*)d_in[8];
  const float* b2 = (const float*)d_in[9];
  const float* ln1_g = (const float*)d_in[10];
  const float* ln1_b = (const float*)d_in[11];
  const float* ln2_g = (const float*)d_in[12];
  const float* ln2_b = (const float*)d_in[13];
  float* out = (float*)d_out;

  char* ws = (char*)d_ws;
  size_t off = 0;
  auto alloc = [&](size_t bytes) -> void* {
    void* p = ws + off;
    off += (bytes + 255) & ~(size_t)255;
    return p;
  };
  const size_t NE = (size_t)NROWS * Dc;
  const size_t WE = (size_t)Dc * Dc;
  __bf16* x_bf   = (__bf16*)alloc(NE * 2);
  __bf16* WtQ    = (__bf16*)alloc(WE * 2);
  __bf16* WtK    = (__bf16*)alloc(WE * 2);
  __bf16* WtV    = (__bf16*)alloc(WE * 2);
  __bf16* WoT    = (__bf16*)alloc(WE * 2);
  __bf16* W1T    = (__bf16*)alloc(WE * 2);
  __bf16* W2T    = (__bf16*)alloc(WE * 2);
  __bf16* qb     = (__bf16*)alloc(NE * 2);
  __bf16* kb     = (__bf16*)alloc(NE * 2);
  __bf16* vTb    = (__bf16*)alloc(NE * 2);
  __bf16* attn   = (__bf16*)alloc(NE * 2);
  float*  y_f    = (float*) alloc(NE * 4);
  float*  x1_f   = (float*) alloc(NE * 4);
  __bf16* x1_bf  = (__bf16*)alloc(NE * 2);
  __bf16* h_bf   = (__bf16*)alloc(NE * 2);
  float*  ff_f   = (float*) alloc(NE * 4);
  float*  mbias  = (float*) alloc((size_t)NROWS * 4);

  f32_to_bf16<<<dim3((int)((NE + 255) / 256)), 256, 0, stream>>>(x, x_bf, (int)NE);
  pack_qkv_w<<<dim3((int)((WE + 255) / 256)), 256, 0, stream>>>(Wq, WtQ);
  pack_qkv_w<<<dim3((int)((WE + 255) / 256)), 256, 0, stream>>>(Wk, WtK);
  pack_qkv_w<<<dim3((int)((WE + 255) / 256)), 256, 0, stream>>>(Wv, WtV);
  transpose_to_bf16<<<dim3((int)((WE + 255) / 256)), 256, 0, stream>>>(Wo, WoT);
  transpose_to_bf16<<<dim3((int)((WE + 255) / 256)), 256, 0, stream>>>(W1, W1T);
  transpose_to_bf16<<<dim3((int)((WE + 255) / 256)), 256, 0, stream>>>(W2, W2T);
  token_bias<<<dim3(NROWS / 256), 256, 0, stream>>>(tokens, mbias, NROWS);

  dim3 ggrid(Dc / 128, NROWS / 64);   // (4, 128)
  gemm_bf16<<<ggrid, 128, 0, stream>>>(x_bf, WtQ, NROWS, Dc, Dc, 0, nullptr, nullptr, qb);
  gemm_bf16<<<ggrid, 128, 0, stream>>>(x_bf, WtK, NROWS, Dc, Dc, 0, nullptr, nullptr, kb);
  gemm_bf16<<<ggrid, 128, 0, stream>>>(x_bf, WtV, NROWS, Dc, Dc, 4, nullptr, nullptr, vTb);

  flash_attn<<<dim3(Bc * Hc * (Sc / 64)), 128, 0, stream>>>(qb, kb, vTb, mbias, attn);

  gemm_bf16<<<ggrid, 128, 0, stream>>>(attn, WoT, NROWS, Dc, Dc, 1, x, nullptr, y_f);
  ln_residual<<<dim3(NROWS), 256, 0, stream>>>(y_f, x, ln1_g, ln1_b, x1_f, x1_bf);
  gemm_bf16<<<ggrid, 128, 0, stream>>>(x1_bf, W1T, NROWS, Dc, Dc, 2, nullptr, b1, h_bf);
  gemm_bf16<<<ggrid, 128, 0, stream>>>(h_bf, W2T, NROWS, Dc, Dc, 3, nullptr, b2, ff_f);
  ln_residual<<<dim3(NROWS), 256, 0, stream>>>(ff_f, x1_f, ln2_g, ln2_b, out, nullptr);
}
